// ParallelTransformerBlock_35673998361108
// MI455X (gfx1250) — matrix operand order repaired and checked by reading, not compiled, not run on hardware
//
#include <hip/hip_runtime.h>

// ---------------------------------------------------------------------------
// ParallelTransformerBlock on MI455X (gfx1250, wave32, WMMA).
// Memory-bound (~236MB fp32 weights vs 23.3TB/s HBM): convert weights to bf16
// once (118MB -> L2-resident), run all GEMMs as bf16 WMMA, flash-attention
// for the multi-query attention. CDNA5 data movement:
//   - GEMM A tiles staged with global_load_async_to_lds_b128 (ASYNCcnt)
//   - attention K tiles staged with the Tensor Data Mover (TENSORcnt)
// ---------------------------------------------------------------------------

typedef unsigned short u16;
typedef __attribute__((ext_vector_type(16))) __bf16 v16bf;
typedef __attribute__((ext_vector_type(8)))  float  floatx8;
typedef __attribute__((ext_vector_type(4)))  unsigned int v4u;
typedef __attribute__((ext_vector_type(8)))  int v8i;
typedef __attribute__((ext_vector_type(4)))  int v4i;

union F16x16 { uint4 u4[2]; v16bf v; };   // one WMMA 16-bit operand (32B/lane)

#define DIM        2048
#define N_SEQ      2048
#define BATCH      2
#define ROWS       (BATCH * N_SEQ)        // 4096
#define HEADS      32
#define DHEAD      64
#define ATTN_INNER 2048
#define FF_INNER   8192
#define FUSED_OUT  18560                  // 2048 + 128 + 2*8192
#define FF_OFF     (ATTN_INNER + 2*DHEAD) // 2176

__device__ __forceinline__ u16 f2bf(float f) {
    unsigned u = __float_as_uint(f);
    u += 0x7fffu + ((u >> 16) & 1u);      // round-to-nearest-even
    return (u16)(u >> 16);
}
__device__ __forceinline__ float bf2f(u16 h) {
    return __uint_as_float(((unsigned)h) << 16);
}

// --- CDNA5 async copy: global -> LDS, per-lane 16B, tracked by ASYNCcnt ----
__device__ __forceinline__ void async_ld_b128(unsigned lds_off,
                                              unsigned long long gaddr) {
    asm volatile("global_load_async_to_lds_b128 %0, %1, off"
                 :: "v"(lds_off), "v"(gaddr) : "memory");
}
__device__ __forceinline__ void wait_async0() {
    asm volatile("s_wait_asynccnt 0x0" ::: "memory");
}

// --- CDNA5 Tensor Data Mover: 2D bf16 tile global -> LDS (TENSORcnt) ------
// rows x 64-element tile from a row-major (nrows_total, 64) bf16 tensor.
__device__ __forceinline__ void tdm_load_tile_2d(unsigned lds_off,
                                                 const u16* gsrc,
                                                 unsigned tile_rows) {
#if __has_builtin(__builtin_amdgcn_tensor_load_to_lds)
    unsigned long long ga = (unsigned long long)(uintptr_t)gsrc;
    v4u g0 = { 1u,                                   // count=1, user mode
               lds_off,                              // lds_addr  (bits 63:32)
               (unsigned)ga,                         // global_addr lo
               (unsigned)((ga >> 32) & 0x01FFFFFFu)  // global_addr hi (56:32)
                   | (2u << 30) };                   // type=2 ("image")
    v8i g1 = { (int)(1u << 16),        // data_size=1 (2 bytes/elem)
               (int)(64u << 16),       // tensor_dim0 = 64      (bits 63:48)
               (int)((unsigned)N_SEQ << 16), // tensor_dim1     (bits 95:80)
               (int)(64u << 16),       // tile_dim0 = 64        (bits 127:112)
               (int)tile_rows,         // tile_dim1             (bits 143:128)
               (int)64,                // tensor_dim0_stride=64 (bits 191:160)
               0, 0 };
    v4i z4 = {};
#if __clang_major__ >= 23
    v8i z8 = {};
    __builtin_amdgcn_tensor_load_to_lds(g0, g1, z4, z4, z8, 0);
#else
    __builtin_amdgcn_tensor_load_to_lds(g0, g1, z4, z4, 0);
#endif
#else
    (void)lds_off; (void)gsrc; (void)tile_rows;
#endif
}

// ---------------------------------------------------------------------------
// fp32 -> bf16 elementwise convert (grid-stride)
// ---------------------------------------------------------------------------
__global__ void convert_bf16_kernel(const float* __restrict__ in,
                                    u16* __restrict__ out, long n) {
    long i = (long)blockIdx.x * blockDim.x + threadIdx.x;
    long stride = (long)gridDim.x * blockDim.x;
    for (; i < n; i += stride) out[i] = f2bf(in[i]);
}

// ---------------------------------------------------------------------------
// LayerNorm: one block per row of 2048, writes bf16
// ---------------------------------------------------------------------------
__global__ __launch_bounds__(256)
void layernorm_kernel(const float* __restrict__ x,
                      const float* __restrict__ lw,
                      const float* __restrict__ lb,
                      u16* __restrict__ xn) {
    __shared__ float red[256];
    const int tid = threadIdx.x;
    const int row = blockIdx.x;
    const float* xr = x + (size_t)row * DIM;

    float v[8], s = 0.f, s2 = 0.f;
#pragma unroll
    for (int e = 0; e < 8; ++e) {
        v[e] = xr[e * 256 + tid];
        s += v[e]; s2 += v[e] * v[e];
    }
    red[tid] = s; __syncthreads();
    for (int st = 128; st > 0; st >>= 1) {
        if (tid < st) red[tid] += red[tid + st];
        __syncthreads();
    }
    float mu = red[0] * (1.0f / DIM);
    __syncthreads();
    red[tid] = s2; __syncthreads();
    for (int st = 128; st > 0; st >>= 1) {
        if (tid < st) red[tid] += red[tid + st];
        __syncthreads();
    }
    float var = red[0] * (1.0f / DIM) - mu * mu;
    float rs = rsqrtf(var + 1e-5f);
#pragma unroll
    for (int e = 0; e < 8; ++e) {
        int col = e * 256 + tid;
        xn[(size_t)row * DIM + col] = f2bf((v[e] - mu) * rs * lw[col] + lb[col]);
    }
}

// ---------------------------------------------------------------------------
// bf16 GEMM: C[M,N] = A[M,K] * B[K,N]  (A,B bf16 row-major, fp32 accumulate)
// Block tile 128x128 (8 waves, each wave 64x32 = 4x2 WMMA frags), K-step 32.
// A tile staged by async copy (ASYNCcnt); B tile transposed via ds_store_b16.
// mode 0: C fp32 store; mode 1: C fp32 accumulate; mode 2: C bf16 store.
// ---------------------------------------------------------------------------
#define BM  128
#define BN  128
#define BK  32
#define LDA 40   // halves per As row (32 + 8 pad, keeps 16B alignment)
#define LDB 40   // halves per Bs column (column-major tile)

__global__ __launch_bounds__(256)
void gemm_bf16_kernel(const u16* __restrict__ A, const u16* __restrict__ B,
                      float* __restrict__ Cf, u16* __restrict__ Cb,
                      int M, int N, int K, int mode) {
    __shared__ __align__(16) u16 As[BM * LDA];
    __shared__ __align__(16) u16 Bs[BN * LDB];

    const int tid  = threadIdx.x;
    const int lane = tid & 31;
    const int wave = tid >> 5;
    const int wm   = (wave >> 2) * 64;  // 0 / 64
    const int wn   = (wave & 3)  * 32;  // 0..96
    const int m0   = blockIdx.y * BM;
    const int n0   = blockIdx.x * BN;
    const int lr   = lane & 15;
    const int hi   = lane >> 4;         // 0 / 1

    const unsigned asA = (unsigned)(uintptr_t)As;

    floatx8 c[4][2];
#pragma unroll
    for (int mi = 0; mi < 4; ++mi)
#pragma unroll
        for (int ni = 0; ni < 2; ++ni) c[mi][ni] = (floatx8){};

    for (int kk = 0; kk < K; kk += BK) {
        // ---- stage A tile: async global->LDS (no VGPR round-trip) ----
#pragma unroll
        for (int i = 0; i < 2; ++i) {
            int t   = tid + i * 256;       // 0..511
            int row = t >> 2;              // 0..127
            int kc  = (t & 3) * 8;
            async_ld_b128(asA + (unsigned)(row * LDA + kc) * 2,
                          (unsigned long long)(uintptr_t)
                              (A + (size_t)(m0 + row) * K + kk + kc));
        }
        // ---- stage B tile (transpose to column-major) ----
#pragma unroll
        for (int i = 0; i < 2; ++i) {
            int t    = tid + i * 256;      // 0..511
            int krow = t >> 4;             // 0..31
            int nc   = (t & 15) * 8;
            uint4 vb = *(const uint4*)(B + (size_t)(kk + krow) * N + n0 + nc);
            const u16* hs = (const u16*)&vb;
#pragma unroll
            for (int u = 0; u < 8; ++u) Bs[(nc + u) * LDB + krow] = hs[u];
        }
        if (kk + BK < K) {  // global_prefetch_b8 for next K-slab
            __builtin_prefetch(A + (size_t)(m0 + (tid >> 2)) * K + kk + BK, 0, 0);
            __builtin_prefetch(B + (size_t)(kk + BK + (tid >> 4)) * N + n0 + (tid & 15) * 8, 0, 0);
        }
        wait_async0();
        __syncthreads();

        // ---- per-wave fragments ----
        F16x16 af[4], bfv[2];
#pragma unroll
        for (int mi = 0; mi < 4; ++mi) {
            const u16* p = As + (wm + mi * 16 + lr) * LDA + hi * 8;
            af[mi].u4[0] = *(const uint4*)p;         // K = kb .. kb+7
            af[mi].u4[1] = *(const uint4*)(p + 16);  // K = kb+16 .. kb+23
        }
#pragma unroll
        for (int ni = 0; ni < 2; ++ni) {
            const u16* p = Bs + (wn + ni * 16 + lr) * LDB + hi * 8;
            bfv[ni].u4[0] = *(const uint4*)p;        // K = kb .. kb+7
            bfv[ni].u4[1] = *(const uint4*)(p + 16); // K = kb+16 .. kb+23
        }
#pragma unroll
        for (int mi = 0; mi < 4; ++mi)
#pragma unroll
            for (int ni = 0; ni < 2; ++ni)
                c[mi][ni] = __builtin_amdgcn_wmma_f32_16x16x32_bf16(
                    false, af[mi].v, false, bfv[ni].v,
                    (short)0, c[mi][ni], false, false);
        __syncthreads();
    }

    // ---- epilogue ----
#pragma unroll
    for (int mi = 0; mi < 4; ++mi)
#pragma unroll
        for (int ni = 0; ni < 2; ++ni)
#pragma unroll
            for (int r = 0; r < 8; ++r) {
                int row = m0 + wm + mi * 16 + r + hi * 8;
                int col = n0 + wn + ni * 16 + lr;
                size_t idx = (size_t)row * N + col;
                float v = c[mi][ni][r];
                if (mode == 0)      Cf[idx] = v;
                else if (mode == 1) Cf[idx] += v;
                else                Cb[idx] = f2bf(v);
            }
}

// ---------------------------------------------------------------------------
// RoPE on Q (+ fold in 1/sqrt(64) scale), relayout to (b*h, n, 64) bf16
// ---------------------------------------------------------------------------
__global__ void rope_q_kernel(const u16* __restrict__ proj, u16* __restrict__ q) {
    long t = (long)blockIdx.x * blockDim.x + threadIdx.x;   // b*h*n*64 threads
    int d = t & 63;
    int i = (int)((t >> 6) & (N_SEQ - 1));
    int h = (int)((t >> 17) & (HEADS - 1));
    int b = (int)(t >> 22);
    const u16* src = proj + (size_t)(b * N_SEQ + i) * FUSED_OUT + h * DHEAD;
    float qv  = bf2f(src[d]);
    float rot = (d < 32) ? -bf2f(src[d + 32]) : bf2f(src[d - 32]);
    float inv_freq = __expf(-(float)(d & 31) * (9.210340372f / 32.0f)); // 10000^-(j/32)
    float ang = (float)i * inv_freq;
    float o = (qv * __cosf(ang) + rot * __sinf(ang)) * 0.125f;
    q[(size_t)((b * HEADS + h) * N_SEQ + i) * DHEAD + d] = f2bf(o);
}

// ---------------------------------------------------------------------------
// RoPE on K + copy V (multi-query: single 64-wide head per (b,i))
// ---------------------------------------------------------------------------
__global__ void rope_kv_kernel(const u16* __restrict__ proj,
                               u16* __restrict__ k, u16* __restrict__ v) {
    long t = (long)blockIdx.x * blockDim.x + threadIdx.x;   // b*n*64 threads
    int d = t & 63;
    int i = (int)((t >> 6) & (N_SEQ - 1));
    int b = (int)(t >> 17);
    const u16* src = proj + (size_t)(b * N_SEQ + i) * FUSED_OUT + ATTN_INNER;
    float kv  = bf2f(src[d]);
    float rot = (d < 32) ? -bf2f(src[d + 32]) : bf2f(src[d - 32]);
    float inv_freq = __expf(-(float)(d & 31) * (9.210340372f / 32.0f));
    float ang = (float)i * inv_freq;
    size_t dst = (size_t)(b * N_SEQ + i) * DHEAD + d;
    k[dst] = f2bf(kv * __cosf(ang) + rot * __sinf(ang));
    v[dst] = src[DHEAD + d];   // V copied verbatim (already bf16)
}

// ---------------------------------------------------------------------------
// SwiGLU: ffact = silu(gate) * val   (bf16 in proj -> bf16)
// ---------------------------------------------------------------------------
__global__ void ff_act_kernel(const u16* __restrict__ proj, u16* __restrict__ ff) {
    long t = (long)blockIdx.x * blockDim.x + threadIdx.x;   // rows*8192 threads
    int  j   = (int)(t & (FF_INNER - 1));
    long row = t >> 13;
    const u16* src = proj + (size_t)row * FUSED_OUT + FF_OFF;
    float val  = bf2f(src[j]);
    float gate = bf2f(src[FF_INNER + j]);
    float silu = gate / (1.0f + __expf(-gate));
    ff[(size_t)row * FF_INNER + j] = f2bf(silu * val);
}

// ---------------------------------------------------------------------------
// Flash attention (multi-query, mask == all-true).
// Block = 128 threads (4 waves); each wave owns 16 q-rows. j streamed in
// chunks of 32. K chunk DMA'd row-major into LDS by the Tensor Data Mover
// (wave 0 issues, s_wait_tensorcnt, barrier publishes); V chunk transposed
// manually; P tile round-trips through LDS (C-layout -> A-layout).
// Output layout: (b*n, h*64+d) bf16 -> feeds out-projection GEMM directly.
// ---------------------------------------------------------------------------
__global__ __launch_bounds__(128)
void attn_kernel(const u16* __restrict__ Q, const u16* __restrict__ Kt,
                 const u16* __restrict__ V, u16* __restrict__ O) {
    __shared__ __align__(16) u16 Ks[32 * 64];      // row-major (j, d), TDM dest
    __shared__ __align__(16) u16 Vs[64 * 40];      // column-major (d, j) padded
    __shared__ __align__(16) u16 Ps[4][16 * 32];   // per-wave P tile

    const int tid  = threadIdx.x;
    const int lane = tid & 31;
    const int wave = tid >> 5;
    const int lr   = lane & 15;
    const int hi   = lane >> 4;
    const int bh   = blockIdx.y;           // 0..63
    const int b    = bh >> 5;
    const int h    = bh & 31;
    const int i0   = blockIdx.x * 64 + wave * 16;
    const unsigned ksOff = (unsigned)(uintptr_t)Ks;

    // Q fragments: 16 rows x 64 (two 16x32 A operands), straight from global
    F16x16 qa[2];
#pragma unroll
    for (int f = 0; f < 2; ++f) {
        const u16* p = Q + (size_t)((bh * N_SEQ) + i0 + lr) * DHEAD + f * 32 + hi * 8;
        qa[f].u4[0] = *(const uint4*)p;
        qa[f].u4[1] = *(const uint4*)(p + 16);
    }

    floatx8 o[4];
#pragma unroll
    for (int f = 0; f < 4; ++f) o[f] = (floatx8){};
    float mrow[8], lrow[8];
#pragma unroll
    for (int r = 0; r < 8; ++r) { mrow[r] = -3.0e38f; lrow[r] = 0.f; }

    for (int j0 = 0; j0 < N_SEQ; j0 += 32) {
        // ---- K chunk via Tensor Data Mover (one wave drives the DMA) ----
#if __has_builtin(__builtin_amdgcn_tensor_load_to_lds)
        if (wave == 0) {
            tdm_load_tile_2d(ksOff, Kt + (size_t)(b * N_SEQ + j0) * DHEAD, 32u);
            __builtin_amdgcn_s_wait_tensorcnt(0);
        }
#else
#pragma unroll
        for (int i = 0; i < 2; ++i) {
            int t  = tid + i * 128;
            int j  = t >> 3;
            int c8 = (t & 7) * 8;
            uint4 kv = *(const uint4*)(Kt + (size_t)(b * N_SEQ + j0 + j) * DHEAD + c8);
            *(uint4*)(Ks + j * 64 + c8) = kv;
        }
#endif
        // ---- V chunk: transpose into column-major LDS ----
#pragma unroll
        for (int i = 0; i < 2; ++i) {
            int t  = tid + i * 128;          // 0..255
            int j  = t >> 3;                 // 0..31
            int c8 = (t & 7) * 8;
            uint4 vv = *(const uint4*)(V + (size_t)(b * N_SEQ + j0 + j) * DHEAD + c8);
            const u16* hs = (const u16*)&vv;
#pragma unroll
            for (int u = 0; u < 8; ++u) Vs[(c8 + u) * 40 + j] = hs[u];
        }
        __syncthreads();

        // ---- S = Q * K^T  (two 16x16 score tiles, k=64 in two steps) ----
        floatx8 s[2];
        s[0] = (floatx8){}; s[1] = (floatx8){};
#pragma unroll
        for (int f = 0; f < 2; ++f)
#pragma unroll
            for (int kf = 0; kf < 2; ++kf) {
                F16x16 kb;
                const u16* p = Ks + (f * 16 + lr) * 64 + kf * 32 + hi * 8;
                kb.u4[0] = *(const uint4*)p;
                kb.u4[1] = *(const uint4*)(p + 16);
                s[f] = __builtin_amdgcn_wmma_f32_16x16x32_bf16(
                    false, qa[kf].v, false, kb.v, (short)0, s[f], false, false);
            }

        // ---- online softmax (rows live across 16 lanes of the C layout) ----
#pragma unroll
        for (int r = 0; r < 8; ++r) {
            float mx = fmaxf(s[0][r], s[1][r]);
#pragma unroll
            for (int d = 1; d < 16; d <<= 1) mx = fmaxf(mx, __shfl_xor(mx, d, 16));
            float mnew  = fmaxf(mrow[r], mx);
            float alpha = __expf(mrow[r] - mnew);
            float p0 = __expf(s[0][r] - mnew);
            float p1 = __expf(s[1][r] - mnew);
            float rs = p0 + p1;
#pragma unroll
            for (int d = 1; d < 16; d <<= 1) rs += __shfl_xor(rs, d, 16);
            lrow[r] = lrow[r] * alpha + rs;
            mrow[r] = mnew;
#pragma unroll
            for (int f = 0; f < 4; ++f) o[f][r] *= alpha;
            Ps[wave][(r + hi * 8) * 32 + lr]      = f2bf(p0);
            Ps[wave][(r + hi * 8) * 32 + 16 + lr] = f2bf(p1);
        }
        __syncthreads();

        // ---- O += P * V  (A from LDS in A-layout, B = V^T columns) ----
        F16x16 pa;
        {
            const u16* pp = &Ps[wave][lr * 32 + hi * 8];
            pa.u4[0] = *(const uint4*)pp;
            pa.u4[1] = *(const uint4*)(pp + 16);
        }
#pragma unroll
        for (int f = 0; f < 4; ++f) {
            F16x16 vb;
            const u16* p = Vs + (f * 16 + lr) * 40 + hi * 8;
            vb.u4[0] = *(const uint4*)p;
            vb.u4[1] = *(const uint4*)(p + 16);
            o[f] = __builtin_amdgcn_wmma_f32_16x16x32_bf16(
                false, pa.v, false, vb.v, (short)0, o[f], false, false);
        }
        __syncthreads();
    }

    // ---- normalize + write (b*n, h*64+d) bf16 ----
#pragma unroll
    for (int r = 0; r < 8; ++r) {
        float inv = 1.0f / lrow[r];
        int row_g = b * N_SEQ + i0 + r + hi * 8;
#pragma unroll
        for (int f = 0; f < 4; ++f) {
            int col = h * DHEAD + f * 16 + lr;
            O[(size_t)row_g * ATTN_INNER + col] = f2bf(o[f][r] * inv);
        }
    }
}

// ---------------------------------------------------------------------------
// host-side orchestration
// ---------------------------------------------------------------------------
extern "C" void kernel_launch(void* const* d_in, const int* in_sizes, int n_in,
                              void* d_out, int out_size, void* d_ws, size_t ws_size,
                              hipStream_t stream) {
    (void)in_sizes; (void)n_in; (void)out_size; (void)ws_size;
    const float* x       = (const float*)d_in[0];
    // d_in[1] = attn_mask (all true) -> ignored
    const float* ln_w    = (const float*)d_in[2];
    const float* ln_b    = (const float*)d_in[3];
    const float* w_fused = (const float*)d_in[4];
    const float* w_attn  = (const float*)d_in[5];
    const float* w_ff    = (const float*)d_in[6];
    float* out = (float*)d_out;

    char* wsp = (char*)d_ws;
    size_t off = 0;
    auto alloc = [&](size_t bytes) -> void* {
        void* p = wsp + off;
        off = (off + bytes + 255) & ~(size_t)255;
        return p;
    };
    u16* xn    = (u16*)alloc((size_t)ROWS * DIM * 2);
    u16* wf_b  = (u16*)alloc((size_t)DIM * FUSED_OUT * 2);
    u16* wa_b  = (u16*)alloc((size_t)ATTN_INNER * DIM * 2);
    u16* wo_b  = (u16*)alloc((size_t)FF_INNER * DIM * 2);
    u16* proj  = (u16*)alloc((size_t)ROWS * FUSED_OUT * 2);
    u16* qb    = (u16*)alloc((size_t)BATCH * HEADS * N_SEQ * DHEAD * 2);
    u16* kb    = (u16*)alloc((size_t)BATCH * N_SEQ * DHEAD * 2);
    u16* vb    = (u16*)alloc((size_t)BATCH * N_SEQ * DHEAD * 2);
    u16* ao    = (u16*)alloc((size_t)ROWS * ATTN_INNER * 2);
    u16* ffa   = (u16*)alloc((size_t)ROWS * FF_INNER * 2);

    // 1) weights -> bf16 (once; resident in 192MB L2 for the GEMMs)
    convert_bf16_kernel<<<4096, 256, 0, stream>>>(w_fused, wf_b, (long)DIM * FUSED_OUT);
    convert_bf16_kernel<<<1024, 256, 0, stream>>>(w_attn,  wa_b, (long)ATTN_INNER * DIM);
    convert_bf16_kernel<<<2048, 256, 0, stream>>>(w_ff,    wo_b, (long)FF_INNER * DIM);

    // 2) LayerNorm
    layernorm_kernel<<<ROWS, 256, 0, stream>>>(x, ln_w, ln_b, xn);

    // 3) fused projection GEMM: (4096 x 2048) x (2048 x 18560) -> bf16
    gemm_bf16_kernel<<<dim3(FUSED_OUT / BN, ROWS / BM), 256, 0, stream>>>(
        xn, wf_b, nullptr, proj, ROWS, FUSED_OUT, DIM, 2);

    // 4) epilogues: RoPE(Q)*scale, RoPE(K), V copy, SwiGLU
    rope_q_kernel<<<(BATCH * HEADS * N_SEQ * DHEAD) / 256, 256, 0, stream>>>(proj, qb);
    rope_kv_kernel<<<(BATCH * N_SEQ * DHEAD) / 256, 256, 0, stream>>>(proj, kb, vb);
    ff_act_kernel<<<((long)ROWS * FF_INNER) / 256, 256, 0, stream>>>(proj, ffa);

    // 5) flash attention (grid: 32 q-row blocks x 64 (b,h) pairs)
    attn_kernel<<<dim3(N_SEQ / 64, BATCH * HEADS), 128, 0, stream>>>(qb, kb, vb, ao);

    // 6) out-projection (writes fp32) then FF-out (accumulates) -> d_out
    gemm_bf16_kernel<<<dim3(DIM / BN, ROWS / BM), 256, 0, stream>>>(
        ao, wa_b, out, nullptr, ROWS, DIM, ATTN_INNER, 0);
    gemm_bf16_kernel<<<dim3(DIM / BN, ROWS / BM), 256, 0, stream>>>(
        ffa, wo_b, out, nullptr, ROWS, DIM, FF_INNER, 1);
}
